// ReftLayer_45303315038554
// MI455X (gfx1250) — matrix operand order repaired and checked by reading, not compiled, not run on hardware
//
#include <hip/hip_runtime.h>

// ---------------------------------------------------------------------------
// Problem constants (from reference): B=4, S=2048, D_IN=4096, D_OUT=4096,
// R=32, N_SEL=16.  GEMM: [8192 x 4096] * [4096 x 4096]^T
// ---------------------------------------------------------------------------
constexpr int Mtot = 8192;   // B * S
constexpr int Ntot = 4096;   // D_OUT
constexpr int Ktot = 4096;   // D_IN
constexpr int SEQ  = 2048;
constexpr int RNK  = 32;
constexpr int NSEL = 16;

typedef __attribute__((ext_vector_type(16))) __bf16 v16bf;
typedef __attribute__((ext_vector_type(8)))  __bf16 v8bf;
typedef __attribute__((ext_vector_type(4)))  __bf16 v4bf;
typedef __attribute__((ext_vector_type(8)))  float  v8f;
typedef __attribute__((ext_vector_type(4)))  float  f32x4;

// Block tile geometry: 256 threads = 8 wave32 waves arranged 2 (M) x 4 (N).
// Each wave computes a 64x32 C tile = 4x2 grid of 16x16 WMMA accumulators.
constexpr int BM = 128;
constexpr int BN = 128;
constexpr int BK = 64;
constexpr int LDS_STRIDE = BK + 8;   // bf16 elems; 144B row stride, 16B aligned
constexpr int TILE_ELEMS = BM * LDS_STRIDE;

__global__ __launch_bounds__(256)
void gemm_bias_bf16(const float* __restrict__ X,     // [Mtot, Ktot]
                    const float* __restrict__ W,     // [Ntot, Ktot]
                    const float* __restrict__ bias,  // [Ntot]
                    float* __restrict__ out)         // [Mtot, Ntot]
{
    // Double-buffered bf16 tiles: 2 * 2 * 128*72*2B = 73.7 KB (of 320 KB WGP LDS)
    __shared__ __bf16 sA[2][TILE_ELEMS];
    __shared__ __bf16 sB[2][TILE_ELEMS];

    const int tid   = threadIdx.x;
    const int lane  = tid & 31;
    const int wave  = tid >> 5;
    const int waveM = wave >> 2;   // 0..1
    const int waveN = wave & 3;    // 0..3
    const int m0    = blockIdx.y * BM;
    const int n0    = blockIdx.x * BN;

    // Per-thread staging slots: 8 float4 from X, 8 float4 from W per K-tile.
    const int ldR = tid >> 1;                 // row 0..127 (2 threads per row)
    const int ldC = (tid & 1) * 8;            // float4 col base: 0 or 8
    const float* gA = X + (size_t)(m0 + ldR) * Ktot + ldC * 4;
    const float* gB = W + (size_t)(n0 + ldR) * Ktot + ldC * 4;

    v8f acc[4][2];
#pragma unroll
    for (int i = 0; i < 4; ++i)
#pragma unroll
        for (int j = 0; j < 2; ++j)
#pragma unroll
            for (int e = 0; e < 8; ++e) acc[i][j][e] = 0.0f;

    const int halfHi = lane >> 4;   // 0 for lanes 0-15, 1 for lanes 16-31
    const int lmod   = lane & 15;

    f32x4 pa[8], pb[8];

    // ---- Prologue: fetch + stage K-tile 0 into buffer 0 ----
#pragma unroll
    for (int i = 0; i < 8; ++i) {
        pa[i] = *(const f32x4*)(gA + i * 4);
        pb[i] = *(const f32x4*)(gB + i * 4);
    }
#pragma unroll
    for (int i = 0; i < 8; ++i) {
        v4bf ha, hb;
#pragma unroll
        for (int e = 0; e < 4; ++e) { ha[e] = (__bf16)pa[i][e]; hb[e] = (__bf16)pb[i][e]; }
        *(v4bf*)&sA[0][ldR * LDS_STRIDE + (ldC + i) * 4] = ha;
        *(v4bf*)&sB[0][ldR * LDS_STRIDE + (ldC + i) * 4] = hb;
    }
    __syncthreads();

    int buf = 0;
    for (int kt = 0; kt < Ktot; kt += BK) {
        const bool hasNext = (kt + BK) < Ktot;

        // ---- Issue global loads for the NEXT K-tile (latency hidden by WMMA) ----
        if (hasNext) {
#pragma unroll
            for (int i = 0; i < 8; ++i) {
                pa[i] = *(const f32x4*)(gA + (kt + BK) + i * 4);
                pb[i] = *(const f32x4*)(gB + (kt + BK) + i * 4);
            }
        }

        // ---- Compute on current buffer: two K=32 WMMA steps ----
        const __bf16* bA = sA[buf];
        const __bf16* bB = sB[buf];
#pragma unroll
        for (int k0 = 0; k0 < BK; k0 += 32) {
            v16bf fa[4];
#pragma unroll
            for (int i = 0; i < 4; ++i) {
                // A layout (16-bit, 16x32): lanes 0-15 hold M=lane, K {0..7,16..23};
                // lanes 16-31 hold same M rows, K {8..15,24..31}.
                const int row = waveM * 64 + i * 16 + lmod;
                const int kb  = k0 + (halfHi << 3);
                union { v16bf v; v8bf h[2]; } u;
                u.h[0] = *(const v8bf*)&bA[row * LDS_STRIDE + kb];
                u.h[1] = *(const v8bf*)&bA[row * LDS_STRIDE + kb + 16];
                fa[i] = u.v;
            }
            v16bf fb[2];
#pragma unroll
            for (int j = 0; j < 2; ++j) {
                // B layout (32x16): lane = N; lanes 0-15 hold K {0..7,16..23},
                // lanes 16-31 hold K {8..15,24..31} (matches A).
                const int col = waveN * 32 + j * 16 + lmod;
                const int kb  = k0 + (halfHi << 3);
                union { v16bf v; v8bf h[2]; } u;
                u.h[0] = *(const v8bf*)&bB[col * LDS_STRIDE + kb];
                u.h[1] = *(const v8bf*)&bB[col * LDS_STRIDE + kb + 16];
                fb[j] = u.v;
            }
#pragma unroll
            for (int i = 0; i < 4; ++i)
#pragma unroll
                for (int j = 0; j < 2; ++j)
                    acc[i][j] = __builtin_amdgcn_wmma_f32_16x16x32_bf16(
                        /*neg_a=*/false, fa[i], /*neg_b=*/false, fb[j],
                        /*c_mod=*/(short)0, acc[i][j],
                        /*reuse_a=*/false, /*reuse_b=*/false);
        }

        // ---- Convert + stage prefetched tile into the alternate buffer ----
        if (hasNext) {
            const int nb = buf ^ 1;
#pragma unroll
            for (int i = 0; i < 8; ++i) {
                v4bf ha, hb;
#pragma unroll
                for (int e = 0; e < 4; ++e) { ha[e] = (__bf16)pa[i][e]; hb[e] = (__bf16)pb[i][e]; }
                *(v4bf*)&sA[nb][ldR * LDS_STRIDE + (ldC + i) * 4] = ha;
                *(v4bf*)&sB[nb][ldR * LDS_STRIDE + (ldC + i) * 4] = hb;
            }
        }
        __syncthreads();
        buf ^= 1;
    }

    // ---- Writeback + fused bias.  C layout: VGPR p -> M = p + 8*halfHi,
    //      lane&15 -> N. ----
    const int mBase = m0 + waveM * 64;
    const int nBase = n0 + waveN * 32;
    float bvals[2];
#pragma unroll
    for (int j = 0; j < 2; ++j) bvals[j] = bias[nBase + j * 16 + lmod];

#pragma unroll
    for (int i = 0; i < 4; ++i) {
        const int mrow = mBase + i * 16 + (halfHi << 3);
#pragma unroll
        for (int j = 0; j < 2; ++j) {
            const int nn = nBase + j * 16 + lmod;
#pragma unroll
            for (int p = 0; p < 8; ++p)
                out[(size_t)(mrow + p) * Ntot + nn] = acc[i][j][p] + bvals[j];
        }
    }
}

// ---------------------------------------------------------------------------
// ReFT epilogue: for each selected token row h (already = W x + b in `out`):
//   t      = (A_w - R_w) h + A_b              (rank 32)
//   h_new  = h + R_w^T t
// One block per (batch, selected-token).  Deterministic two-stage reduction
// (no FP atomics), so graph replays produce bit-identical output.
// ---------------------------------------------------------------------------
__global__ __launch_bounds__(256)
void reft_fix(float* __restrict__ out,
              const float* __restrict__ Rw,   // [RNK, Ntot]
              const float* __restrict__ Aw,   // [RNK, Ntot]
              const float* __restrict__ Ab,   // [RNK]
              const int*   __restrict__ loc)  // [B, NSEL]
{
    __shared__ float sel[Ntot];     // 16 KB
    __shared__ float part[256];
    __shared__ float t[RNK];

    const int b   = blockIdx.x >> 4;
    const int n   = blockIdx.x & 15;
    const int pos = loc[b * NSEL + n];
    float* row = out + ((size_t)b * SEQ + pos) * Ntot;

    for (int d = threadIdx.x; d < Ntot; d += 256) sel[d] = row[d];
    __syncthreads();

    // t[r] = Ab[r] + sum_d (Aw[r][d] - Rw[r][d]) * sel[d]
    const int r = threadIdx.x >> 3;   // 0..31
    const int j = threadIdx.x & 7;    // 8 partial accumulators per r
    float partial = 0.0f;
    for (int d = j; d < Ntot; d += 8)
        partial += (Aw[(size_t)r * Ntot + d] - Rw[(size_t)r * Ntot + d]) * sel[d];
    part[threadIdx.x] = partial;
    __syncthreads();

    if (threadIdx.x < RNK) {
        float s = Ab[threadIdx.x];
#pragma unroll
        for (int q = 0; q < 8; ++q) s += part[threadIdx.x * 8 + q];
        t[threadIdx.x] = s;
    }
    __syncthreads();

    // row[d] = sel[d] + sum_r Rw[r][d] * t[r]
    for (int d = threadIdx.x; d < Ntot; d += 256) {
        float off = 0.0f;
#pragma unroll
        for (int rr = 0; rr < RNK; ++rr)
            off += Rw[(size_t)rr * Ntot + d] * t[rr];
        row[d] = sel[d] + off;
    }
}

extern "C" void kernel_launch(void* const* d_in, const int* in_sizes, int n_in,
                              void* d_out, int out_size, void* d_ws, size_t ws_size,
                              hipStream_t stream)
{
    (void)in_sizes; (void)n_in; (void)out_size; (void)d_ws; (void)ws_size;

    const float* x      = (const float*)d_in[0];   // [B,S,D_IN]
    const float* W_base = (const float*)d_in[1];   // [D_OUT,D_IN]
    const float* b_base = (const float*)d_in[2];   // [D_OUT]
    const float* R_w    = (const float*)d_in[3];   // [R,D_OUT]
    const float* A_w    = (const float*)d_in[4];   // [R,D_OUT]
    const float* A_b    = (const float*)d_in[5];   // [R]
    const int*   loc    = (const int*)  d_in[6];   // [B,NSEL]
    float* out = (float*)d_out;                    // [B,S,D_OUT]

    dim3 grid(Ntot / BN, Mtot / BM);   // (32, 64)
    gemm_bias_bf16<<<grid, 256, 0, stream>>>(x, W_base, b_base, out);
    reft_fix<<<4 * NSEL, 256, 0, stream>>>(out, R_w, A_w, A_b, loc);
}
